// PolicyNetwork_21784074125383
// MI455X (gfx1250) — matrix operand order repaired and checked by reading, not compiled, not run on hardware
//
#include <hip/hip_runtime.h>

// ---------------------------------------------------------------------------
// Mamba forward (24 layers) for MI455X / gfx1250, wave32 + WMMA bf16.
// GEMMs run on v_wmma_f32_16x16x32_bf16; weights are pre-transposed and
// converted f32->bf16 into workspace so each WMMA fragment is loaded with
// two contiguous 16-byte global loads per lane.
// ---------------------------------------------------------------------------

typedef __attribute__((ext_vector_type(16))) __bf16        v16bf;
typedef __attribute__((ext_vector_type(8)))  float         v8f;
typedef __attribute__((ext_vector_type(4)))  unsigned int  u32x4;

#define L_SEQ   1024
#define D_MODEL 768
#define D_INNER 1536
#define N_LAYER 24
#define D_STATE 16
#define D_CONVW 4
#define DT_RANK 48

__device__ __forceinline__ unsigned short f2bf(float f) {
    unsigned int u = __float_as_uint(f);
    unsigned int r = u + 0x7FFFu + ((u >> 16) & 1u);   // round-to-nearest-even
    return (unsigned short)(r >> 16);
}

union Frag { u32x4 u[2]; v16bf v; };

// ---------------------------------------------------------------------------
// GEMM: C(MxN, f32) = A(Mx Kpad, bf16 row-major) * Bt(Npad x Kpad, bf16 = B^T)
// Block = 256 threads = 8 waves arranged 4(M) x 2(N); wave tile = 32x64
// (2x4 grid of 16x16 WMMA tiles). M is a multiple of 128; Npad, Kpad padded.
// epi: 0 = store, 1 = C += acc, 2 = C = softplus(acc + bias[n])
// ---------------------------------------------------------------------------
__global__ __launch_bounds__(256)
void k_gemm_bf16(const unsigned short* __restrict__ A,
                 const unsigned short* __restrict__ Bt,
                 float* __restrict__ C,
                 int N, int Kpad, int epi, const float* __restrict__ bias)
{
    const int tid  = threadIdx.x;
    const int wid  = tid >> 5;
    const int lane = tid & 31;
    const int wm   = wid & 3;      // wave row   (0..3)
    const int wn   = wid >> 2;     // wave col   (0..1)
    const int lm   = lane & 15;    // lane index within half-wave
    const int hi   = lane >> 4;    // half-wave select

    const int row0 = blockIdx.y * 128 + wm * 32;
    const int col0 = blockIdx.x * 128 + wn * 64;

    v8f acc[2][4];
#pragma unroll
    for (int i = 0; i < 2; ++i)
#pragma unroll
        for (int j = 0; j < 4; ++j) acc[i][j] = (v8f)0.0f;

    for (int kk = 0; kk < Kpad; kk += 32) {
        Frag a[2], b[4];
        const int ka = kk + hi * 8;        // A lane layout: v0..3 -> k, v4..7 -> k+16
#pragma unroll
        for (int mt = 0; mt < 2; ++mt) {
            const unsigned short* ap = A + (size_t)(row0 + mt * 16 + lm) * Kpad + ka;
            a[mt].u[0] = *(const u32x4*)(ap);
            a[mt].u[1] = *(const u32x4*)(ap + 16);
        }
        const int kb = kk + hi * 8;        // B lane layout: v0..3 -> k, v4..7 -> k+16
#pragma unroll
        for (int nt = 0; nt < 4; ++nt) {
            const unsigned short* bp = Bt + (size_t)(col0 + nt * 16 + lm) * Kpad + kb;
            b[nt].u[0] = *(const u32x4*)(bp);
            b[nt].u[1] = *(const u32x4*)(bp + 16);
        }
#pragma unroll
        for (int mt = 0; mt < 2; ++mt)
#pragma unroll
            for (int nt = 0; nt < 4; ++nt)
                acc[mt][nt] = __builtin_amdgcn_wmma_f32_16x16x32_bf16(
                    false, a[mt].v, false, b[nt].v, (short)0, acc[mt][nt], false, false);
    }

    // Epilogue: C/D layout -> lane = col, VGPR r = row (r + 8*hi)
#pragma unroll
    for (int mt = 0; mt < 2; ++mt) {
#pragma unroll
        for (int nt = 0; nt < 4; ++nt) {
            const int n = col0 + nt * 16 + lm;
            if (n < N) {
                const int mbase = row0 + mt * 16 + hi * 8;
#pragma unroll
                for (int r = 0; r < 8; ++r) {
                    float v = acc[mt][nt][r];
                    const size_t o = (size_t)(mbase + r) * N + n;
                    if (epi == 1)      C[o] += v;
                    else if (epi == 2) {
                        v += bias[n];
                        C[o] = (v > 20.f) ? v : log1pf(__expf(v));
                    } else             C[o] = v;
                }
            }
        }
    }
}

// Convert f32 weight W (K x N row-major) into bf16 W^T (Npad x Kpad), zero pad.
__global__ void k_convert_wT(const float* __restrict__ src, unsigned short* __restrict__ dst,
                             int K, int N, int Kpad, int Npad)
{
    int idx = blockIdx.x * 256 + threadIdx.x;
    if (idx >= Kpad * Npad) return;
    int n = idx / Kpad, k = idx - n * Kpad;
    float v = (n < N && k < K) ? src[(size_t)k * N + n] : 0.f;
    dst[idx] = f2bf(v);
}

// u = x @ w_enc + b_enc
__global__ void k_encode(const float* __restrict__ x, const float* __restrict__ we,
                         const float* __restrict__ be, float* __restrict__ u)
{
    int idx = blockIdx.x * 256 + threadIdx.x;
    if (idx >= L_SEQ * D_MODEL) return;
    int t = idx / D_MODEL, d = idx - t * D_MODEL;
    float acc = be[d];
#pragma unroll
    for (int i = 0; i < 4; ++i) acc += x[t * 4 + i] * we[i * D_MODEL + d];
    u[idx] = acc;
}

// RMSNorm row of u -> bf16 activations for the in_proj GEMM.
__global__ __launch_bounds__(256)
void k_rmsnorm_bf16(const float* __restrict__ u, const float* __restrict__ w,
                    unsigned short* __restrict__ hbf)
{
    __shared__ float red[256];
    const int t = blockIdx.x;
    const float* row = u + (size_t)t * D_MODEL;
    float s = 0.f;
    for (int d = threadIdx.x; d < D_MODEL; d += 256) { float v = row[d]; s += v * v; }
    red[threadIdx.x] = s; __syncthreads();
    for (int st = 128; st > 0; st >>= 1) {
        if (threadIdx.x < st) red[threadIdx.x] += red[threadIdx.x + st];
        __syncthreads();
    }
    const float scale = rsqrtf(red[0] / (float)D_MODEL + 1e-5f);
    for (int d = threadIdx.x; d < D_MODEL; d += 256)
        hbf[(size_t)t * D_MODEL + d] = f2bf(row[d] * scale * w[d]);
}

// Depthwise causal conv (width 4) + bias + SiLU; emit f32 and bf16 copies.
__global__ void k_conv_silu(const float* __restrict__ xz, const float* __restrict__ cw,
                            const float* __restrict__ cb, float* __restrict__ xc,
                            unsigned short* __restrict__ xcbf)
{
    int idx = blockIdx.x * 256 + threadIdx.x;
    if (idx >= L_SEQ * D_INNER) return;
    int t = idx / D_INNER, c = idx - t * D_INNER;
    float acc = cb[c];
#pragma unroll
    for (int j = 0; j < D_CONVW; ++j) {
        int tt = t - (D_CONVW - 1) + j;
        if (tt >= 0) acc += cw[c * D_CONVW + j] * xz[(size_t)tt * (2 * D_INNER) + c];
    }
    float s = acc / (1.f + __expf(-acc));
    xc[idx]   = s;
    xcbf[idx] = f2bf(s);
}

// Extract dt-rank slice of dbl into a zero-padded bf16 A matrix (Kpad = 64).
__global__ void k_dt_in(const float* __restrict__ dbl, unsigned short* __restrict__ dtA)
{
    int idx = blockIdx.x * 256 + threadIdx.x;
    if (idx >= L_SEQ * 64) return;
    int t = idx >> 6, k = idx & 63;
    float v = (k < DT_RANK) ? dbl[(size_t)t * 80 + k] : 0.f;
    dtA[idx] = f2bf(v);
}

// Selective scan: one thread per channel, 16 states in registers,
// per-timestep B/C (16+16 floats) broadcast through LDS.
__global__ __launch_bounds__(256)
void k_scan(const float* __restrict__ dt, const float* __restrict__ dbl,
            const float* __restrict__ xc, const float* __restrict__ xz,
            const float* __restrict__ A_log, const float* __restrict__ Dsk,
            unsigned short* __restrict__ ybf)
{
    __shared__ float sBC[32];
    const int c = blockIdx.x * 256 + threadIdx.x;   // 6 blocks * 256 = 1536
    float Ac[D_STATE], h[D_STATE];
#pragma unroll
    for (int j = 0; j < D_STATE; ++j) {
        Ac[j] = -__expf(A_log[(size_t)c * D_STATE + j]);
        h[j]  = 0.f;
    }
    const float Dv = Dsk[c];
    for (int t = 0; t < L_SEQ; ++t) {
        if (threadIdx.x < 32) sBC[threadIdx.x] = dbl[(size_t)t * 80 + DT_RANK + threadIdx.x];
        __syncthreads();
        const float dtv = dt[(size_t)t * D_INNER + c];
        const float xv  = xc[(size_t)t * D_INNER + c];
        float y = 0.f;
#pragma unroll
        for (int j = 0; j < D_STATE; ++j) {
            const float dA = __expf(dtv * Ac[j]);
            h[j] = dA * h[j] + dtv * sBC[j] * xv;
            y += h[j] * sBC[D_STATE + j];
        }
        const float zv = xz[(size_t)t * (2 * D_INNER) + D_INNER + c];
        const float o  = (y + Dv * xv) * (zv / (1.f + __expf(-zv)));
        ybf[(size_t)t * D_INNER + c] = f2bf(o);
        __syncthreads();
    }
}

// Final RMSNorm of last row + 768x2 head.
__global__ __launch_bounds__(256)
void k_head(const float* __restrict__ u, const float* __restrict__ nfw,
            const float* __restrict__ fcw, const float* __restrict__ fcb,
            float* __restrict__ out)
{
    __shared__ float red[256];
    const float* row = u + (size_t)(L_SEQ - 1) * D_MODEL;
    float s = 0.f;
    for (int d = threadIdx.x; d < D_MODEL; d += 256) { float v = row[d]; s += v * v; }
    red[threadIdx.x] = s; __syncthreads();
    for (int st = 128; st > 0; st >>= 1) {
        if (threadIdx.x < st) red[threadIdx.x] += red[threadIdx.x + st];
        __syncthreads();
    }
    const float scale = rsqrtf(red[0] / (float)D_MODEL + 1e-5f);
    __syncthreads();
    float p0 = 0.f, p1 = 0.f;
    for (int d = threadIdx.x; d < D_MODEL; d += 256) {
        float v = row[d] * scale * nfw[d];
        p0 += v * fcw[d * 2 + 0];
        p1 += v * fcw[d * 2 + 1];
    }
    red[threadIdx.x] = p0; __syncthreads();
    for (int st = 128; st > 0; st >>= 1) {
        if (threadIdx.x < st) red[threadIdx.x] += red[threadIdx.x + st];
        __syncthreads();
    }
    const float r0 = red[0]; __syncthreads();
    red[threadIdx.x] = p1; __syncthreads();
    for (int st = 128; st > 0; st >>= 1) {
        if (threadIdx.x < st) red[threadIdx.x] += red[threadIdx.x + st];
        __syncthreads();
    }
    if (threadIdx.x == 0) { out[0] = r0 + fcb[0]; out[1] = red[0] + fcb[1]; }
}

// ---------------------------------------------------------------------------
extern "C" void kernel_launch(void* const* d_in, const int* in_sizes, int n_in,
                              void* d_out, int out_size, void* d_ws, size_t ws_size,
                              hipStream_t stream)
{
    (void)in_sizes; (void)n_in; (void)out_size; (void)ws_size;
    const float* x        = (const float*)d_in[0];
    const float* w_enc    = (const float*)d_in[1];
    const float* b_enc    = (const float*)d_in[2];
    const float* norm_w   = (const float*)d_in[3];
    const float* in_proj  = (const float*)d_in[4];
    const float* conv_w   = (const float*)d_in[5];
    const float* conv_b   = (const float*)d_in[6];
    const float* x_proj   = (const float*)d_in[7];
    const float* dt_proj  = (const float*)d_in[8];
    const float* dt_bias  = (const float*)d_in[9];
    const float* A_log    = (const float*)d_in[10];
    const float* D_skip   = (const float*)d_in[11];
    const float* out_proj = (const float*)d_in[12];
    const float* norm_f   = (const float*)d_in[13];
    const float* fc_w     = (const float*)d_in[14];
    const float* fc_b     = (const float*)d_in[15];

    // Workspace carve-up (256B aligned).
    char*  base = (char*)d_ws;
    size_t off  = 0;
    auto alloc = [&](size_t bytes) -> void* {
        void* p = base + off;
        off = (off + bytes + 255) & ~(size_t)255;
        return p;
    };
    float*          u     = (float*)         alloc((size_t)L_SEQ * D_MODEL * 4);
    unsigned short* hbf   = (unsigned short*)alloc((size_t)L_SEQ * D_MODEL * 2);
    unsigned short* WtIn  = (unsigned short*)alloc((size_t)(2 * D_INNER) * D_MODEL * 2);
    float*          xz    = (float*)         alloc((size_t)L_SEQ * 2 * D_INNER * 4);
    float*          xc    = (float*)         alloc((size_t)L_SEQ * D_INNER * 4);
    unsigned short* xcbf  = (unsigned short*)alloc((size_t)L_SEQ * D_INNER * 2);
    unsigned short* WtXp  = (unsigned short*)alloc((size_t)128 * D_INNER * 2);
    float*          dbl   = (float*)         alloc((size_t)L_SEQ * 80 * 4);
    unsigned short* dtA   = (unsigned short*)alloc((size_t)L_SEQ * 64 * 2);
    unsigned short* WtDt  = (unsigned short*)alloc((size_t)D_INNER * 64 * 2);
    float*          dtb   = (float*)         alloc((size_t)L_SEQ * D_INNER * 4);
    unsigned short* ybf   = (unsigned short*)alloc((size_t)L_SEQ * D_INNER * 2);
    unsigned short* WtOut = (unsigned short*)alloc((size_t)D_MODEL * D_INNER * 2);

    const dim3 blk(256);
    auto nblk = [](size_t n) { return dim3((unsigned)((n + 255) / 256)); };

    k_encode<<<nblk((size_t)L_SEQ * D_MODEL), blk, 0, stream>>>(x, w_enc, b_enc, u);

    for (int l = 0; l < N_LAYER; ++l) {
        // Weight prep: f32 -> bf16 transposed, padded.
        k_convert_wT<<<nblk((size_t)(2 * D_INNER) * D_MODEL), blk, 0, stream>>>(
            in_proj + (size_t)l * D_MODEL * 2 * D_INNER, WtIn,
            D_MODEL, 2 * D_INNER, D_MODEL, 2 * D_INNER);
        k_convert_wT<<<nblk((size_t)128 * D_INNER), blk, 0, stream>>>(
            x_proj + (size_t)l * D_INNER * 80, WtXp, D_INNER, 80, D_INNER, 128);
        k_convert_wT<<<nblk((size_t)D_INNER * 64), blk, 0, stream>>>(
            dt_proj + (size_t)l * DT_RANK * D_INNER, WtDt, DT_RANK, D_INNER, 64, D_INNER);
        k_convert_wT<<<nblk((size_t)D_MODEL * D_INNER), blk, 0, stream>>>(
            out_proj + (size_t)l * D_INNER * D_MODEL, WtOut, D_INNER, D_MODEL, D_INNER, D_MODEL);

        // h = rmsnorm(u) -> bf16
        k_rmsnorm_bf16<<<dim3(L_SEQ), blk, 0, stream>>>(u, norm_w + (size_t)l * D_MODEL, hbf);

        // xz = h @ in_proj   (1024 x 3072, K = 768)
        k_gemm_bf16<<<dim3(24, 8), blk, 0, stream>>>(hbf, WtIn, xz, 2 * D_INNER, D_MODEL, 0, nullptr);

        // xc = silu(causal_conv(x_part) + b)
        k_conv_silu<<<nblk((size_t)L_SEQ * D_INNER), blk, 0, stream>>>(
            xz, conv_w + (size_t)l * D_INNER * D_CONVW, conv_b + (size_t)l * D_INNER, xc, xcbf);

        // dbl = xc @ x_proj   (1024 x 80, K = 1536)
        k_gemm_bf16<<<dim3(1, 8), blk, 0, stream>>>(xcbf, WtXp, dbl, 80, D_INNER, 0, nullptr);

        // dt = softplus(dbl[:, :48] @ dt_proj + dt_bias)   (1024 x 1536, K = 64 padded)
        k_dt_in<<<nblk((size_t)L_SEQ * 64), blk, 0, stream>>>(dbl, dtA);
        k_gemm_bf16<<<dim3(12, 8), blk, 0, stream>>>(dtA, WtDt, dtb, D_INNER, 64, 2,
                                                     dt_bias + (size_t)l * D_INNER);

        // Selective scan + gating -> bf16 y
        k_scan<<<dim3(D_INNER / 256), blk, 0, stream>>>(
            dtb, dbl, xc, xz, A_log + (size_t)l * D_INNER * D_STATE,
            D_skip + (size_t)l * D_INNER, ybf);

        // u += y @ out_proj   (1024 x 768, K = 1536)
        k_gemm_bf16<<<dim3(6, 8), blk, 0, stream>>>(ybf, WtOut, u, D_MODEL, D_INNER, 1, nullptr);
    }

    k_head<<<dim3(1), blk, 0, stream>>>(u, norm_f, fc_w, fc_b, (float*)d_out);
}
